// PolyGCBaseModel_47253230191370
// MI455X (gfx1250) — matrix operand order repaired and checked by reading, not compiled, not run on hardware
//
#include <hip/hip_runtime.h>
#include <hip/hip_bf16.h>

typedef __attribute__((ext_vector_type(16))) __bf16 v16bf;
typedef __attribute__((ext_vector_type(8)))  float  v8f;

#define FDIM 128
#define LNUM 3

// -------- epilogue/feature flags for the fused WMMA GEMM --------
enum {
  F_DUAL   = 1,    // two A matrices, two B matrices, summed into one acc
  F_SCALEA = 2,    // per-row scale on A0 (deg^-1 / cnt^-1)
  F_BIAS   = 4,
  F_WPLUS  = 8,    // += w[row] * wrow[col]  (129th input row of W_in)
  F_RESID  = 16,   // += resid[row,col]
  F_ROWSUB = 32,   // -= sub[batch[row], col]
  F_ELU    = 64,
  F_LN     = 128,  // row LayerNorm (128 features), fused via shfl reductions
};

__device__ inline float halfsum16(float x) {
  // sum across the 16-lane half (xor bits 0..3 never cross bit 4)
  x += __shfl_xor(x, 1);
  x += __shfl_xor(x, 2);
  x += __shfl_xor(x, 4);
  x += __shfl_xor(x, 8);
  return x;
}

__device__ inline v16bf load_a_frag(const float* base, float scale) {
  // base = &A[row*128 + kblk*32 + koff]; layout per ISA 16-bit A 16x32:
  // lane<16: K = {0..7, 16..23}+koff0 ; lane>=16: koff=8 -> {8..15, 24..31}
  float4 p0 = *(const float4*)(base + 0);
  float4 p1 = *(const float4*)(base + 4);
  float4 p2 = *(const float4*)(base + 16);
  float4 p3 = *(const float4*)(base + 20);
  float f[16] = {p0.x, p0.y, p0.z, p0.w, p1.x, p1.y, p1.z, p1.w,
                 p2.x, p2.y, p2.z, p2.w, p3.x, p3.y, p3.z, p3.w};
  v16bf a;
#pragma unroll
  for (int j = 0; j < 16; ++j) a[j] = (__bf16)(f[j] * scale);
  return a;
}

// Fused 128-col GEMM: out[r, :] = epilogue( A0[r,:]@B0 (+ A1[r,:]@B1) )
// One workgroup = 8 waves = 128 rows; each wave owns 16 rows x 128 cols.
template <int FL>
__global__ __launch_bounds__(256) void gemm128(
    const float* __restrict__ A0, const float* __restrict__ A1,
    const __bf16* __restrict__ B0, const __bf16* __restrict__ B1,
    const float* __restrict__ rowScale, const float* __restrict__ bias,
    const float* __restrict__ wvec, const float* __restrict__ wrow,
    const float* __restrict__ resid, const int* __restrict__ batchIdx,
    const float* __restrict__ sub, const float* __restrict__ lng,
    const float* __restrict__ lnb, float* __restrict__ out, int nrows) {
  constexpr bool DUAL = (FL & F_DUAL) != 0;
  __shared__ __bf16 sB[(DUAL ? 2 : 1) * 16384];

  // cooperative stage of packed weights into LDS (32KB each)
  {
    const uint4* g0 = (const uint4*)B0;
    uint4* s = (uint4*)sB;
    for (int i = threadIdx.x; i < 2048; i += 256) s[i] = g0[i];
    if constexpr (DUAL) {
      const uint4* g1 = (const uint4*)B1;
      for (int i = threadIdx.x; i < 2048; i += 256) s[2048 + i] = g1[i];
    }
  }
  __syncthreads();

  const int lane = threadIdx.x & 31;
  const int wave = threadIdx.x >> 5;
  const int rowBase = blockIdx.x * 128 + wave * 16;

  // ---- A-side addressing (A layout lane mapping) ----
  const int arow = rowBase + (lane & 15);
  const int ar = arow < nrows ? arow : nrows - 1;
  const int koff = (lane & 16) ? 8 : 0;
  const float s0 = (FL & F_SCALEA) ? rowScale[ar] : 1.0f;
  const float* a0p = A0 + (size_t)ar * FDIM + koff;

  v8f acc[8];
#pragma unroll
  for (int t = 0; t < 8; ++t)
#pragma unroll
    for (int v = 0; v < 8; ++v) acc[t][v] = 0.0f;

  const v16bf* sB0 = (const v16bf*)sB;
#pragma unroll
  for (int kblk = 0; kblk < 4; ++kblk) {
    v16bf a0 = load_a_frag(a0p + kblk * 32, s0);
    v16bf a1;
    if constexpr (DUAL) {
      const float* a1p = A1 + (size_t)ar * FDIM + koff;
      a1 = load_a_frag(a1p + kblk * 32, 1.0f);
    }
#pragma unroll
    for (int nblk = 0; nblk < 8; ++nblk) {
      v16bf b0 = sB0[(kblk * 8 + nblk) * 32 + lane];
      acc[nblk] = __builtin_amdgcn_wmma_f32_16x16x32_bf16(
          false, a0, false, b0, (short)0, acc[nblk], false, false);
      if constexpr (DUAL) {
        const v16bf* sB1 = (const v16bf*)(sB + 16384);
        v16bf b1 = sB1[(kblk * 8 + nblk) * 32 + lane];
        acc[nblk] = __builtin_amdgcn_wmma_f32_16x16x32_bf16(
            false, a1, false, b1, (short)0, acc[nblk], false, false);
      }
    }
  }

  // ---- epilogue (C layout: col = t*16 + lane&15, row = rbase + v) ----
  const int colLane = lane & 15;
  const int rbase = rowBase + ((lane >> 4) << 3);

  if constexpr ((FL & F_BIAS) != 0) {
#pragma unroll
    for (int t = 0; t < 8; ++t) {
      float bc = bias[t * 16 + colLane];
#pragma unroll
      for (int v = 0; v < 8; ++v) acc[t][v] += bc;
    }
  }
  if constexpr ((FL & F_WPLUS) != 0) {
    float wv[8];
#pragma unroll
    for (int v = 0; v < 8; ++v) {
      int r = rbase + v;
      wv[v] = wvec[r < nrows ? r : nrows - 1];
    }
#pragma unroll
    for (int t = 0; t < 8; ++t) {
      float wl = wrow[t * 16 + colLane];
#pragma unroll
      for (int v = 0; v < 8; ++v) acc[t][v] += wv[v] * wl;
    }
  }
  if constexpr ((FL & F_RESID) != 0) {
#pragma unroll
    for (int v = 0; v < 8; ++v) {
      int r = rbase + v;
      int rr = r < nrows ? r : nrows - 1;
      const float* rp = resid + (size_t)rr * FDIM + colLane;
#pragma unroll
      for (int t = 0; t < 8; ++t) acc[t][v] += rp[t * 16];
    }
  }
  if constexpr ((FL & F_ROWSUB) != 0) {
#pragma unroll
    for (int v = 0; v < 8; ++v) {
      int r = rbase + v;
      int rr = r < nrows ? r : nrows - 1;
      int bg = batchIdx[rr];
      const float* sp = sub + (size_t)bg * FDIM + colLane;
#pragma unroll
      for (int t = 0; t < 8; ++t) acc[t][v] -= sp[t * 16];
    }
  }
  if constexpr ((FL & F_ELU) != 0) {
#pragma unroll
    for (int t = 0; t < 8; ++t)
#pragma unroll
      for (int v = 0; v < 8; ++v) {
        float xv = acc[t][v];
        acc[t][v] = xv > 0.0f ? xv : __expf(xv) - 1.0f;
      }
  }
  if constexpr ((FL & F_LN) != 0) {
    float mean[8], rstd[8];
#pragma unroll
    for (int v = 0; v < 8; ++v) {
      float s = 0.0f;
#pragma unroll
      for (int t = 0; t < 8; ++t) s += acc[t][v];
      mean[v] = halfsum16(s) * (1.0f / 128.0f);
    }
#pragma unroll
    for (int v = 0; v < 8; ++v) {
      float s = 0.0f;
#pragma unroll
      for (int t = 0; t < 8; ++t) {
        float d = acc[t][v] - mean[v];
        s += d * d;
      }
      rstd[v] = rsqrtf(halfsum16(s) * (1.0f / 128.0f) + 1e-5f);
    }
#pragma unroll
    for (int t = 0; t < 8; ++t) {
      float g = lng[t * 16 + colLane];
      float bb = lnb[t * 16 + colLane];
#pragma unroll
      for (int v = 0; v < 8; ++v)
        acc[t][v] = (acc[t][v] - mean[v]) * rstd[v] * g + bb;
    }
  }
#pragma unroll
  for (int v = 0; v < 8; ++v) {
    int r = rbase + v;
    if (r < nrows) {
      float* op = out + (size_t)r * FDIM + colLane;
#pragma unroll
      for (int t = 0; t < 8; ++t) op[t * 16] = acc[t][v];
    }
  }
}

// ---- repack a row-major 128x128 fp32 (optionally sum of two) weight into
//      the WMMA B-fragment bf16 layout (lane-contiguous 32B fragments) ----
__global__ void k_repack(const float* __restrict__ W0,
                         const float* __restrict__ W1, __bf16* __restrict__ dst) {
  int idx = blockIdx.x * 256 + threadIdx.x;  // 16384 total
  int j = idx & 15;
  int lane = (idx >> 4) & 31;
  int nblk = (idx >> 9) & 7;
  int kblk = (idx >> 12) & 3;
  int n = nblk * 16 + (lane & 15);
  int K = kblk * 32 + ((lane & 16) ? 8 : 0) + (j & 7) + ((j & 8) ? 16 : 0);
  float v = W0[K * FDIM + n];
  if (W1) v += W1[K * FDIM + n];
  dst[idx] = (__bf16)v;
}

// ---- graph primitives ----
__global__ void k_deg(const int* __restrict__ dst, float* __restrict__ deg, int E) {
  int e = blockIdx.x * blockDim.x + threadIdx.x;
  if (e < E) atomicAdd(&deg[dst[e]], 1.0f);
}
__global__ void k_cnt(const int* __restrict__ batch, float* __restrict__ cnt, int N) {
  int i = blockIdx.x * blockDim.x + threadIdx.x;
  if (i < N) atomicAdd(&cnt[batch[i]], 1.0f);
}
__global__ void k_recip(float* __restrict__ a, int n) {
  int i = blockIdx.x * blockDim.x + threadIdx.x;
  if (i < n) a[i] = 1.0f / fmaxf(a[i], 1.0f);
}
__global__ void k_scatter_edges(const int* __restrict__ src, const int* __restrict__ dst,
                                const float* __restrict__ h, float* __restrict__ agg, int E) {
  long long t = (long long)blockIdx.x * blockDim.x + threadIdx.x;
  int e = (int)(t >> 5);
  if (e >= E) return;
  int lane = (int)(t & 31);
  int s = src[e], d = dst[e];
  float4 v = *(const float4*)(h + (size_t)s * FDIM + lane * 4);
  float* p = agg + (size_t)d * FDIM + lane * 4;
  atomicAdd(p + 0, v.x);
  atomicAdd(p + 1, v.y);
  atomicAdd(p + 2, v.z);
  atomicAdd(p + 3, v.w);
}
__global__ void k_scatter_nodes(const int* __restrict__ batch, const float* __restrict__ h,
                                float* __restrict__ outp, int N) {
  long long t = (long long)blockIdx.x * blockDim.x + threadIdx.x;
  int i = (int)(t >> 5);
  if (i >= N) return;
  int lane = (int)(t & 31);
  int b = batch[i];
  float4 v = *(const float4*)(h + (size_t)i * FDIM + lane * 4);
  float* p = outp + (size_t)b * FDIM + lane * 4;
  atomicAdd(p + 0, v.x);
  atomicAdd(p + 1, v.y);
  atomicAdd(p + 2, v.z);
  atomicAdd(p + 3, v.w);
}

// mean-pool LayerNorm head: hh[b] = LN(p[b] * cntinv[b])
__global__ __launch_bounds__(256) void k_pool_ln(const float* __restrict__ p,
                                                 const float* __restrict__ cinv,
                                                 const float* __restrict__ g,
                                                 const float* __restrict__ b,
                                                 float* __restrict__ out) {
  int wave = threadIdx.x >> 5, lane = threadIdx.x & 31;
  int row = blockIdx.x * 8 + wave;
  float c = cinv[row];
  float4 x = *(const float4*)(p + (size_t)row * FDIM + lane * 4);
  x.x *= c; x.y *= c; x.z *= c; x.w *= c;
  float s = x.x + x.y + x.z + x.w;
  s += __shfl_xor(s, 1); s += __shfl_xor(s, 2); s += __shfl_xor(s, 4);
  s += __shfl_xor(s, 8); s += __shfl_xor(s, 16);
  float m = s * (1.0f / 128.0f);
  float dx = x.x - m, dy = x.y - m, dz = x.z - m, dw = x.w - m;
  float v = dx * dx + dy * dy + dz * dz + dw * dw;
  v += __shfl_xor(v, 1); v += __shfl_xor(v, 2); v += __shfl_xor(v, 4);
  v += __shfl_xor(v, 8); v += __shfl_xor(v, 16);
  float rs = rsqrtf(v * (1.0f / 128.0f) + 1e-5f);
  int c0 = lane * 4;
  float4 o;
  o.x = dx * rs * g[c0 + 0] + b[c0 + 0];
  o.y = dy * rs * g[c0 + 1] + b[c0 + 1];
  o.z = dz * rs * g[c0 + 2] + b[c0 + 2];
  o.w = dw * rs * g[c0 + 3] + b[c0 + 3];
  *(float4*)(out + (size_t)row * FDIM + c0) = o;
}

extern "C" void kernel_launch(void* const* d_in, const int* in_sizes, int n_in,
                              void* d_out, int out_size, void* d_ws, size_t ws_size,
                              hipStream_t stream) {
  const float* x       = (const float*)d_in[0];
  const float* w       = (const float*)d_in[1];
  const int*   ei      = (const int*)d_in[2];
  const int*   batch   = (const int*)d_in[3];
  const float* W_in    = (const float*)d_in[4];
  const float* b_in    = (const float*)d_in[5];
  const float* sage_Wl = (const float*)d_in[6];
  const float* sage_bl = (const float*)d_in[7];
  const float* sage_Wr = (const float*)d_in[8];
  const float* ln1_g   = (const float*)d_in[9];
  const float* ln1_b   = (const float*)d_in[10];
  const float* lin1_W  = (const float*)d_in[11];
  const float* lin1_b  = (const float*)d_in[12];
  const float* lin2_W  = (const float*)d_in[13];
  const float* lin2_b  = (const float*)d_in[14];
  const float* ln2_g   = (const float*)d_in[15];
  const float* ln2_b   = (const float*)d_in[16];
  const float* mdf_W   = (const float*)d_in[17];
  const float* mdf_b   = (const float*)d_in[18];
  const float* pln1_g  = (const float*)d_in[19];
  const float* pln1_b  = (const float*)d_in[20];
  const float* plin1_W = (const float*)d_in[21];
  const float* plin1_b = (const float*)d_in[22];
  const float* plin2_W = (const float*)d_in[23];
  const float* plin2_b = (const float*)d_in[24];
  const float* pln2_g  = (const float*)d_in[25];
  const float* pln2_b  = (const float*)d_in[26];

  const int N = in_sizes[0] / FDIM;      // 100000
  const int E = in_sizes[2] / 2;         // 1600000
  const int G = out_size / FDIM;         // 1024
  const int* esrc = ei;
  const int* edst = ei + E;
  const size_t NB = (size_t)N * FDIM * sizeof(float);
  const size_t GB = (size_t)G * FDIM * sizeof(float);
  const int MM = FDIM * FDIM;            // 16384

  size_t off = 0;
  auto carve = [&](size_t bytes) -> char* {
    char* p = (char*)d_ws + off;
    off += (bytes + 255) & ~(size_t)255;
    return p;
  };
  float*  bufA   = (float*)carve(NB);       // h
  float*  bufB   = (float*)carve(NB);       // agg / t / h2
  float*  bufC   = (float*)carve(NB);       // y1
  float*  deginv = (float*)carve((size_t)N * 4);
  float*  cntinv = (float*)carve((size_t)G * 4);
  float*  gm     = (float*)carve(GB);       // also reused as pooled t2
  float*  gmrow  = (float*)carve(GB);
  float*  pbuf   = (float*)carve(GB);
  float*  hh     = (float*)carve(GB);
  __bf16* pk     = (__bf16*)carve((size_t)17 * MM * sizeof(__bf16));
  auto PK = [&](int i) { return pk + (size_t)i * MM; };

  const int TPB = 256;
  const int nodeBlocks = (N + 127) / 128;
  const int gBlocks = (G + 127) / 128;

  // degrees / counts (recomputed each call; deterministic)
  hipMemsetAsync(deginv, 0, (size_t)N * 4, stream);
  hipMemsetAsync(cntinv, 0, (size_t)G * 4, stream);
  k_deg<<<(E + TPB - 1) / TPB, TPB, 0, stream>>>(edst, deginv, E);
  k_cnt<<<(N + TPB - 1) / TPB, TPB, 0, stream>>>(batch, cntinv, N);
  k_recip<<<(N + TPB - 1) / TPB, TPB, 0, stream>>>(deginv, N);
  k_recip<<<(G + TPB - 1) / TPB, TPB, 0, stream>>>(cntinv, G);

  // weight repack to bf16 WMMA B-fragments
  k_repack<<<64, TPB, 0, stream>>>(W_in, nullptr, PK(0));          // W_in rows 0..127
  for (int l = 0; l < LNUM; ++l) {
    k_repack<<<64, TPB, 0, stream>>>(sage_Wl + l * MM, nullptr, PK(1 + l));
    k_repack<<<64, TPB, 0, stream>>>(sage_Wr + l * MM, nullptr, PK(4 + l));
    k_repack<<<64, TPB, 0, stream>>>(lin1_W + l * MM, nullptr, PK(7 + l));
    k_repack<<<64, TPB, 0, stream>>>(lin2_W + l * MM, nullptr, PK(10 + l));
  }
  k_repack<<<64, TPB, 0, stream>>>(mdf_W, mdf_W + MM, PK(13));     // Wtop+Wbot
  k_repack<<<64, TPB, 0, stream>>>(mdf_W + MM, nullptr, PK(14));   // Wbot
  k_repack<<<64, TPB, 0, stream>>>(plin1_W, nullptr, PK(15));
  k_repack<<<64, TPB, 0, stream>>>(plin2_W, nullptr, PK(16));

  // input linear: h = x@W_in[0:128] + w*W_in[128,:] + b_in
  gemm128<F_BIAS | F_WPLUS><<<nodeBlocks, TPB, 0, stream>>>(
      x, nullptr, PK(0), nullptr, nullptr, b_in, w, W_in + 128 * FDIM,
      nullptr, nullptr, nullptr, nullptr, nullptr, bufA, N);

  // GC blocks
  for (int l = 0; l < LNUM; ++l) {
    hipMemsetAsync(bufB, 0, NB, stream);
    k_scatter_edges<<<(int)(((long long)E * 32 + TPB - 1) / TPB), TPB, 0, stream>>>(
        esrc, edst, bufA, bufB, E);
    // y1 = LN( (agg/deg)@Wl + bl + h@Wr + h )
    gemm128<F_DUAL | F_SCALEA | F_BIAS | F_RESID | F_LN><<<nodeBlocks, TPB, 0, stream>>>(
        bufB, bufA, PK(1 + l), PK(4 + l), deginv, sage_bl + l * FDIM,
        nullptr, nullptr, bufA, nullptr, nullptr, ln1_g + l * FDIM,
        ln1_b + l * FDIM, bufC, N);
    // t = elu(y1@W1 + bb1)
    gemm128<F_BIAS | F_ELU><<<nodeBlocks, TPB, 0, stream>>>(
        bufC, nullptr, PK(7 + l), nullptr, nullptr, lin1_b + l * FDIM,
        nullptr, nullptr, nullptr, nullptr, nullptr, nullptr, nullptr, bufB, N);
    // h = LN(t@W2 + bb2 + y1)
    gemm128<F_BIAS | F_RESID | F_LN><<<nodeBlocks, TPB, 0, stream>>>(
        bufB, nullptr, PK(10 + l), nullptr, nullptr, lin2_b + l * FDIM,
        nullptr, nullptr, bufC, nullptr, nullptr, ln2_g + l * FDIM,
        ln2_b + l * FDIM, bufA, N);
  }

  // MeanDiff: h2 = h@(Wtop+Wbot) + mdf_b - (gm_mean@Wbot)[batch]
  hipMemsetAsync(gm, 0, GB, stream);
  k_scatter_nodes<<<(int)(((long long)N * 32 + TPB - 1) / TPB), TPB, 0, stream>>>(
      batch, bufA, gm, N);
  gemm128<F_SCALEA><<<gBlocks, TPB, 0, stream>>>(
      gm, nullptr, PK(14), nullptr, cntinv, nullptr, nullptr, nullptr,
      nullptr, nullptr, nullptr, nullptr, nullptr, gmrow, G);
  gemm128<F_BIAS | F_ROWSUB><<<nodeBlocks, TPB, 0, stream>>>(
      bufA, nullptr, PK(13), nullptr, nullptr, mdf_b, nullptr, nullptr,
      nullptr, batch, gmrow, nullptr, nullptr, bufB, N);

  // mean pooling head
  hipMemsetAsync(pbuf, 0, GB, stream);
  k_scatter_nodes<<<(int)(((long long)N * 32 + TPB - 1) / TPB), TPB, 0, stream>>>(
      batch, bufB, pbuf, N);
  k_pool_ln<<<G / 8, TPB, 0, stream>>>(pbuf, cntinv, pln1_g, pln1_b, hh);
  // t2 = elu(hh@plin1 + b)
  gemm128<F_BIAS | F_ELU><<<gBlocks, TPB, 0, stream>>>(
      hh, nullptr, PK(15), nullptr, nullptr, plin1_b, nullptr, nullptr,
      nullptr, nullptr, nullptr, nullptr, nullptr, gm, G);
  // out = LN(t2@plin2 + b + hh)
  gemm128<F_BIAS | F_RESID | F_LN><<<gBlocks, TPB, 0, stream>>>(
      gm, nullptr, PK(16), nullptr, nullptr, plin2_b, nullptr, nullptr,
      hh, nullptr, nullptr, pln2_g, pln2_b, (float*)d_out, G);
}
